// CrossModalAttention_51299089383488
// MI455X (gfx1250) — matrix operand order repaired and checked by reading, not compiled, not run on hardware
//
#include <hip/hip_runtime.h>
#include <hip/hip_bf16.h>
#include <math.h>

typedef _Float16 half_t;
typedef __attribute__((ext_vector_type(16))) _Float16 v16h;
typedef __attribute__((ext_vector_type(8)))  _Float16 v8h;
typedef __attribute__((ext_vector_type(8)))  float    v8f;

// ---------------------------------------------------------------------------
// Problem constants (reference: n=384, h=32, m=50, GRID=5, ORDER=3, BASIS=8)
// Grid knots: g_t = (t-3)*0.4 - 1, t = 0..11  (uniform, h = 0.4)
// ---------------------------------------------------------------------------
#define N_TOK  384
#define HDIM   32
#define MDIM   50
#define NB     8
#define FEAT9  9            // [silu, B0..B7]
#define KCOLS  288          // 32 * 9
#define NPAD   112          // 100 outputs (SA:50 | SB:50) padded to 7*16
#define GH     0.4f

__device__ __forceinline__ float silu_f(float x) {
    return x / (1.0f + __expf(-x));
}

// Exact Cox-de Boor recursion (matches reference bit-for-bit structure).
__device__ __forceinline__ void bspline8_dense(float x, float* B) {
    float b0[11];
#pragma unroll
    for (int t = 0; t < 11; ++t) {
        float g0 = (t - 3) * GH - 1.0f;
        float g1 = (t - 2) * GH - 1.0f;
        b0[t] = (x >= g0 && x < g1) ? 1.0f : 0.0f;
    }
    float b1[10];
#pragma unroll
    for (int t = 0; t < 10; ++t) {
        float gt  = (t - 3) * GH - 1.0f;
        float gt2 = (t - 1) * GH - 1.0f;
        b1[t] = (x - gt) * 2.5f * b0[t] + (gt2 - x) * 2.5f * b0[t + 1];
    }
    float b2[9];
#pragma unroll
    for (int t = 0; t < 9; ++t) {
        float gt  = (t - 3) * GH - 1.0f;
        float gt3 = (t    ) * GH - 1.0f;
        b2[t] = (x - gt) * 1.25f * b1[t] + (gt3 - x) * 1.25f * b1[t + 1];
    }
#pragma unroll
    for (int t = 0; t < 8; ++t) {
        float gt  = (t - 3) * GH - 1.0f;
        float gt4 = (t + 1) * GH - 1.0f;
        B[t] = (x - gt) * (1.0f / 1.2f) * b2[t] + (gt4 - x) * (1.0f / 1.2f) * b2[t + 1];
    }
}

// ---------------------------------------------------------------------------
// K1a: per-element features F[mod][i][k*9 + {0..8}] = [silu(q), B0..B7] (f16)
// ---------------------------------------------------------------------------
__global__ void build_feat_kernel(const float* __restrict__ x,
                                  const float* __restrict__ y,
                                  const float* __restrict__ tg,
                                  half_t* __restrict__ F) {
    int t = blockIdx.x * blockDim.x + threadIdx.x;
    if (t >= 3 * N_TOK * HDIM) return;
    int mod = t / (N_TOK * HDIM);
    int rem = t % (N_TOK * HDIM);
    int i = rem / HDIM, k = rem % HDIM;
    const float* q = (mod == 0) ? x : (mod == 1) ? y : tg;
    float v = q[i * HDIM + k];
    half_t* out = F + ((size_t)mod * N_TOK + i) * KCOLS + k * FEAT9;
    out[0] = (half_t)silu_f(v);
    float B[8];
    bspline8_dense(v, B);
#pragma unroll
    for (int b = 0; b < 8; ++b) out[1 + b] = (half_t)B[b];
}

// ---------------------------------------------------------------------------
// K1b: pack layer-1 weights, N-major transposed: WhT[mod][n][col] (f16)
//   n<50  -> SA output o=n    (uses bw1[o, k],     sw1[o, k,     b])
//   n<100 -> SB output o=n-50 (uses bw1[o, 32+k],  sw1[o, 32+k,  b])
//   n>=100 -> zero padding
// ---------------------------------------------------------------------------
__global__ void pack_w_kernel(const float* __restrict__ x1bw, const float* __restrict__ x1sw,
                              const float* __restrict__ y1bw, const float* __restrict__ y1sw,
                              const float* __restrict__ t1bw, const float* __restrict__ t1sw,
                              half_t* __restrict__ WhT) {
    int t = blockIdx.x * blockDim.x + threadIdx.x;
    if (t >= 3 * NPAD * KCOLS) return;
    int mod = t / (NPAD * KCOLS);
    int rem = t % (NPAD * KCOLS);
    int n = rem / KCOLS, col = rem % KCOLS;
    int k = col / FEAT9, c = col % FEAT9;
    const float* bw = (mod == 0) ? x1bw : (mod == 1) ? y1bw : t1bw;
    const float* sw = (mod == 0) ? x1sw : (mod == 1) ? y1sw : t1sw;
    float v = 0.0f;
    if (n < 100) {
        int o    = (n < 50) ? n : n - 50;
        int koff = (n < 50) ? k : 32 + k;
        v = (c == 0) ? bw[o * 64 + koff] : sw[(o * 64 + koff) * 8 + (c - 1)];
    }
    WhT[t] = (half_t)v;   // flat index order == [mod][n][col]
}

// ---------------------------------------------------------------------------
// K2: WMMA GEMM  SASB[mod][384][112] = F[mod][384][288] @ W[mod][288][112]
// One wave32 per 16x16 output tile, v_wmma_f32_16x16x32_f16, 9 K-steps.
// CDNA5 wave32 layouts (ISA 7.12.2):
//   A lane r(<16): e0..7 = A[r][hw*8 + 0..7],  e8..15 = A[r][16 + hw*8 + 0..7]
//   B lane n(<16): e0..7 = B[hw*8 + 0..7][n],  e8..15 = B[16 + hw*8 + 0..7][n]   (so W pre-transposed N-major)
//   D VGPR v:      D[v + 8*hw][n]
// ---------------------------------------------------------------------------
__global__ __launch_bounds__(32) void gemm_sasb_kernel(const half_t* __restrict__ F,
                                                       const half_t* __restrict__ WhT,
                                                       float* __restrict__ SASB) {
    int tm = blockIdx.x, tn = blockIdx.y, mod = blockIdx.z;
    int lane = threadIdx.x;
    int hw = lane >> 4, r = lane & 15;
    const half_t* Fm = F   + (size_t)mod * N_TOK * KCOLS;
    const half_t* Wm = WhT + (size_t)mod * NPAD  * KCOLS;
    const half_t* arow = Fm + (size_t)(tm * 16 + r) * KCOLS;
    const half_t* brow = Wm + (size_t)(tn * 16 + r) * KCOLS;
    v8f acc = {};
#pragma unroll
    for (int kk = 0; kk < 9; ++kk) {
        int k0 = kk * 32;
        v8h alo = *(const v8h*)(arow + k0 + hw * 8);
        v8h ahi = *(const v8h*)(arow + k0 + 16 + hw * 8);
        v8h blo = *(const v8h*)(brow + k0 + hw * 8);
        v8h bhi = *(const v8h*)(brow + k0 + 16 + hw * 8);
        v16h a, b;
#pragma unroll
        for (int e = 0; e < 8; ++e) {
            a[e] = alo[e]; a[e + 8] = ahi[e];
            b[e] = blo[e]; b[e + 8] = bhi[e];
        }
        acc = __builtin_amdgcn_wmma_f32_16x16x32_f16(false, a, false, b,
                                                     (short)0, acc, false, false);
    }
#pragma unroll
    for (int v = 0; v < 8; ++v) {
        int row = tm * 16 + v + 8 * hw;
        int col = tn * 16 + r;
        SASB[((size_t)mod * N_TOK + row) * NPAD + col] = acc[v];
    }
}

// ---------------------------------------------------------------------------
// K3: hot kernel. logits[mod][i][j] = sum_o silu(z)*bw2[o] + sum_b B_b(z)*sw2[o][b]
// with z = SA[i][o] + SB[j][o]. Closed-form cubic B-spline: 4 active bases.
// 16x16 pair tile per 256-thread block; SA/SB tiles + weights in LDS.
// ---------------------------------------------------------------------------
__global__ __launch_bounds__(256) void pair_att_kernel(
    const float* __restrict__ SASB,
    const float* __restrict__ x2bw, const float* __restrict__ x2sw,
    const float* __restrict__ y2bw, const float* __restrict__ y2sw,
    const float* __restrict__ t2bw, const float* __restrict__ t2sw,
    float* __restrict__ logits) {
    __shared__ float SAs[MDIM][16];
    __shared__ float SBs[MDIM][16];
    __shared__ float sw2s[MDIM * NB];
    __shared__ float bw2s[MDIM];

    int mod = blockIdx.z;
    const float* bw2 = (mod == 0) ? x2bw : (mod == 1) ? y2bw : t2bw;
    const float* sw2 = (mod == 0) ? x2sw : (mod == 1) ? y2sw : t2sw;
    int tid = threadIdx.x;
    int i0 = blockIdx.x * 16, j0 = blockIdx.y * 16;
    const float* base = SASB + (size_t)mod * N_TOK * NPAD;

    for (int t = tid; t < MDIM * 16; t += 256) {
        int o = t >> 4, r = t & 15;
        SAs[o][r] = base[(size_t)(i0 + r) * NPAD + o];
        SBs[o][r] = base[(size_t)(j0 + r) * NPAD + 50 + o];
    }
    for (int t = tid; t < MDIM * NB; t += 256) sw2s[t] = sw2[t];
    if (tid < MDIM) bw2s[tid] = bw2[tid];
    __syncthreads();

    int ti = tid >> 4, tj = tid & 15;
    float acc = 0.0f;
#pragma unroll 2
    for (int o = 0; o < MDIM; ++o) {
        float z = SAs[o][ti] + SBs[o][tj];
        acc = fmaf(silu_f(z), bw2s[o], acc);
        // cell coordinate: xc = (z - (-1 - 3h)) / h ; valid support is [0, 11)
        float xc = (z + 2.2f) * 2.5f;
        if (xc > 0.0f && xc < 11.0f) {
            float cf = floorf(xc);
            int   c  = (int)cf;
            float u  = xc - cf;
            float u2 = u * u, u3 = u2 * u;
            float om = 1.0f - u;
            float wgt[4];
            wgt[0] = om * om * om * (1.0f / 6.0f);                       // b = c-3
            wgt[1] = (3.0f * u3 - 6.0f * u2 + 4.0f) * (1.0f / 6.0f);     // b = c-2
            wgt[2] = (-3.0f * u3 + 3.0f * u2 + 3.0f * u + 1.0f) * (1.0f / 6.0f); // b=c-1
            wgt[3] = u3 * (1.0f / 6.0f);                                 // b = c
#pragma unroll
            for (int t4 = 0; t4 < 4; ++t4) {
                int b = c - 3 + t4;
                bool valid = (b >= 0) && (b < NB);
                int bb = valid ? b : 0;
                float wv = valid ? wgt[t4] : 0.0f;
                acc = fmaf(wv, sw2s[o * NB + bb], acc);
            }
        }
    }
    logits[((size_t)mod * N_TOK + (i0 + ti)) * N_TOK + (j0 + tj)] = acc;
}

// ---------------------------------------------------------------------------
// K4: dynamic fusion weights. feats = [mean(x), mean(y), mean(t)] (96),
// kan2 with f1/f2, softmax -> w[3]. Single block.
// ---------------------------------------------------------------------------
__global__ __launch_bounds__(128) void fusion_w_kernel(
    const float* __restrict__ x, const float* __restrict__ y, const float* __restrict__ tg,
    const float* __restrict__ f1bw, const float* __restrict__ f1sw,
    const float* __restrict__ f2bw, const float* __restrict__ f2sw,
    float* __restrict__ wout) {
    __shared__ float f9[96 * FEAT9];
    __shared__ float z1[MDIM];
    __shared__ float z2[3];
    int tid = threadIdx.x;
    if (tid < 96) {
        int which = tid / 32, cc = tid % 32;
        const float* q = (which == 0) ? x : (which == 1) ? y : tg;
        float s = 0.0f;
        for (int r = 0; r < N_TOK; ++r) s += q[r * HDIM + cc];
        float m = s * (1.0f / (float)N_TOK);
        f9[tid * FEAT9] = silu_f(m);
        float B[8];
        bspline8_dense(m, B);
#pragma unroll
        for (int b = 0; b < 8; ++b) f9[tid * FEAT9 + 1 + b] = B[b];
    }
    __syncthreads();
    if (tid < MDIM) {
        float acc = 0.0f;
        for (int c = 0; c < 96; ++c) {
            acc = fmaf(f9[c * FEAT9], f1bw[tid * 96 + c], acc);
#pragma unroll
            for (int b = 0; b < 8; ++b)
                acc = fmaf(f9[c * FEAT9 + 1 + b], f1sw[(tid * 96 + c) * 8 + b], acc);
        }
        z1[tid] = acc;
    }
    __syncthreads();
    if (tid < 3) {
        float acc = 0.0f;
        for (int o = 0; o < MDIM; ++o) {
            float z = z1[o];
            acc = fmaf(silu_f(z), f2bw[tid * MDIM + o], acc);
            float B[8];
            bspline8_dense(z, B);
#pragma unroll
            for (int b = 0; b < 8; ++b)
                acc = fmaf(B[b], f2sw[(tid * MDIM + o) * 8 + b], acc);
        }
        z2[tid] = acc;
    }
    __syncthreads();
    if (tid == 0) {
        float m = fmaxf(z2[0], fmaxf(z2[1], z2[2]));
        float e0 = __expf(z2[0] - m), e1 = __expf(z2[1] - m), e2 = __expf(z2[2] - m);
        float inv = 1.0f / (e0 + e1 + e2);
        wout[0] = e0 * inv; wout[1] = e1 * inv; wout[2] = e2 * inv;
    }
}

// ---------------------------------------------------------------------------
// K5: per-row fused attention: mix logits with w + bias, softmax over j,
// then row of (P @ target). One block per row i.
// ---------------------------------------------------------------------------
__global__ __launch_bounds__(128) void attn_out_kernel(
    const float* __restrict__ logits, const float* __restrict__ wfus,
    const float* __restrict__ bias,   const float* __restrict__ tg,
    float* __restrict__ tatt) {
    __shared__ float p[N_TOK];
    __shared__ float red[128];
    __shared__ float acc2[4][32];
    int i = blockIdx.x, tid = threadIdx.x;
    float w0 = wfus[0], w1 = wfus[1], w2 = wfus[2], bs = bias[0];
    const float* L0 = logits + (size_t)i * N_TOK;
    const float* L1 = logits + ((size_t)N_TOK     + i) * N_TOK;
    const float* L2 = logits + ((size_t)2 * N_TOK + i) * N_TOK;

    float lm = -3.4e38f;
    for (int j = tid; j < N_TOK; j += 128) {
        float v = w0 * L0[j] + w1 * L1[j] + w2 * L2[j] + bs;
        p[j] = v;
        lm = fmaxf(lm, v);
    }
    red[tid] = lm;
    __syncthreads();
    for (int s = 64; s > 0; s >>= 1) {
        if (tid < s) red[tid] = fmaxf(red[tid], red[tid + s]);
        __syncthreads();
    }
    float m = red[0];
    __syncthreads();

    float ls = 0.0f;
    for (int j = tid; j < N_TOK; j += 128) {
        float e = __expf(p[j] - m);
        p[j] = e;
        ls += e;
    }
    red[tid] = ls;
    __syncthreads();
    for (int s = 64; s > 0; s >>= 1) {
        if (tid < s) red[tid] += red[tid + s];
        __syncthreads();
    }
    float inv = 1.0f / red[0];

    int c = tid & 31, ch = tid >> 5;  // 32 cols x 4 j-chunks
    float a = 0.0f;
    for (int j = ch * 96; j < (ch + 1) * 96; ++j) a = fmaf(p[j], tg[j * HDIM + c], a);
    acc2[ch][c] = a;
    __syncthreads();
    if (tid < 32)
        tatt[i * HDIM + tid] =
            (acc2[0][tid] + acc2[1][tid] + acc2[2][tid] + acc2[3][tid]) * inv;
}

// ---------------------------------------------------------------------------
// K6: y2 = relu(kan(relu(kan(target_att, l1)), l2)). One block (1 wave) per row.
// ---------------------------------------------------------------------------
__global__ __launch_bounds__(32) void kan_head_kernel(
    const float* __restrict__ tatt,
    const float* __restrict__ l1bw, const float* __restrict__ l1sw,
    const float* __restrict__ l2bw, const float* __restrict__ l2sw,
    float* __restrict__ out) {
    __shared__ float f9[HDIM * FEAT9];
    __shared__ float y1[HDIM];
    int i = blockIdx.x, tid = threadIdx.x;
    {
        float v = tatt[i * HDIM + tid];
        f9[tid * FEAT9] = silu_f(v);
        float B[8];
        bspline8_dense(v, B);
#pragma unroll
        for (int b = 0; b < 8; ++b) f9[tid * FEAT9 + 1 + b] = B[b];
    }
    __syncthreads();
    {
        float acc = 0.0f;
        for (int k = 0; k < HDIM; ++k) {
            acc = fmaf(f9[k * FEAT9], l1bw[tid * HDIM + k], acc);
#pragma unroll
            for (int b = 0; b < 8; ++b)
                acc = fmaf(f9[k * FEAT9 + 1 + b], l1sw[(tid * HDIM + k) * 8 + b], acc);
        }
        y1[tid] = fmaxf(acc, 0.0f);
    }
    __syncthreads();
    {
        float v = y1[tid];
        f9[tid * FEAT9] = silu_f(v);
        float B[8];
        bspline8_dense(v, B);
#pragma unroll
        for (int b = 0; b < 8; ++b) f9[tid * FEAT9 + 1 + b] = B[b];
    }
    __syncthreads();
    {
        float acc = 0.0f;
        for (int k = 0; k < HDIM; ++k) {
            acc = fmaf(f9[k * FEAT9], l2bw[tid * HDIM + k], acc);
#pragma unroll
            for (int b = 0; b < 8; ++b)
                acc = fmaf(f9[k * FEAT9 + 1 + b], l2sw[(tid * HDIM + k) * 8 + b], acc);
        }
        out[i * HDIM + tid] = fmaxf(acc, 0.0f);
    }
}

// ---------------------------------------------------------------------------
extern "C" void kernel_launch(void* const* d_in, const int* in_sizes, int n_in,
                              void* d_out, int out_size, void* d_ws, size_t ws_size,
                              hipStream_t stream) {
    const float* x    = (const float*)d_in[0];
    const float* y    = (const float*)d_in[1];
    const float* tg   = (const float*)d_in[2];
    const float* bias = (const float*)d_in[3];
    const float* x1bw = (const float*)d_in[4];
    const float* x1sw = (const float*)d_in[5];
    const float* x2bw = (const float*)d_in[6];
    const float* x2sw = (const float*)d_in[7];
    const float* y1bw = (const float*)d_in[8];
    const float* y1sw = (const float*)d_in[9];
    const float* y2bw = (const float*)d_in[10];
    const float* y2sw = (const float*)d_in[11];
    const float* t1bw = (const float*)d_in[12];
    const float* t1sw = (const float*)d_in[13];
    const float* t2bw = (const float*)d_in[14];
    const float* t2sw = (const float*)d_in[15];
    const float* f1bw = (const float*)d_in[16];
    const float* f1sw = (const float*)d_in[17];
    const float* f2bw = (const float*)d_in[18];
    const float* f2sw = (const float*)d_in[19];
    const float* l1bw = (const float*)d_in[20];
    const float* l1sw = (const float*)d_in[21];
    const float* l2bw = (const float*)d_in[22];
    const float* l2sw = (const float*)d_in[23];

    char* ws = (char*)d_ws;
    size_t off = 0;
    auto take = [&](size_t bytes) {
        size_t p = off;
        off += (bytes + 255) & ~(size_t)255;
        return p;
    };
    half_t* F    = (half_t*)(ws + take((size_t)3 * N_TOK * KCOLS * sizeof(half_t)));
    half_t* WhT  = (half_t*)(ws + take((size_t)3 * NPAD  * KCOLS * sizeof(half_t)));
    float*  SASB = (float*) (ws + take((size_t)3 * N_TOK * NPAD  * sizeof(float)));
    float*  LG   = (float*) (ws + take((size_t)3 * N_TOK * N_TOK * sizeof(float)));
    float*  WF   = (float*) (ws + take(16));
    float*  TATT = (float*) (ws + take((size_t)N_TOK * HDIM * sizeof(float)));
    (void)ws_size; (void)n_in; (void)in_sizes; (void)out_size;

    build_feat_kernel<<<(3 * N_TOK * HDIM + 255) / 256, 256, 0, stream>>>(x, y, tg, F);
    pack_w_kernel<<<(3 * NPAD * KCOLS + 255) / 256, 256, 0, stream>>>(
        x1bw, x1sw, y1bw, y1sw, t1bw, t1sw, WhT);
    gemm_sasb_kernel<<<dim3(N_TOK / 16, NPAD / 16, 3), 32, 0, stream>>>(F, WhT, SASB);
    pair_att_kernel<<<dim3(N_TOK / 16, N_TOK / 16, 3), 256, 0, stream>>>(
        SASB, x2bw, x2sw, y2bw, y2sw, t2bw, t2sw, LG);
    fusion_w_kernel<<<1, 128, 0, stream>>>(x, y, tg, f1bw, f1sw, f2bw, f2sw, WF);
    attn_out_kernel<<<N_TOK, 128, 0, stream>>>(LG, WF, bias, tg, TATT);
    kan_head_kernel<<<N_TOK, 32, 0, stream>>>(TATT, l1bw, l1sw, l2bw, l2sw, (float*)d_out);
}
